// GINEClassifier_25933012533301
// MI455X (gfx1250) — compile-verified, matrix operand order checked by reading, not run on hardware
//
#include <hip/hip_runtime.h>
#include <math.h>

#define B_   64
#define N_   1024
#define E_   8192
#define H_   128
#define L_   5
#define FD_  34
#define HC_  193
#define NC_  9
#define FUS_ 256

typedef __attribute__((ext_vector_type(16))) __bf16 v16bf;
typedef __attribute__((ext_vector_type(8)))  float  v8f;

__device__ __forceinline__ unsigned short f32_to_bf16_rne(float f) {
  unsigned int u = __builtin_bit_cast(unsigned int, f);
  unsigned int r = u + 0x7FFFu + ((u >> 16) & 1u);
  return (unsigned short)(r >> 16);
}

// ---------------- utility kernels ----------------

__global__ void k_zero(float* __restrict__ p, long long n) {
  long long i = (long long)blockIdx.x * blockDim.x + threadIdx.x;
  long long st = (long long)gridDim.x * blockDim.x;
  for (; i < n; i += st) p[i] = 0.0f;
}

// pad rows: src (M x K) -> dst (M x Kpad), zero-filled tail
__global__ void k_pad_rows(const float* __restrict__ src, float* __restrict__ dst,
                           int K, int Kpad, long long M) {
  long long total = M * (long long)Kpad;
  long long i = (long long)blockIdx.x * blockDim.x + threadIdx.x;
  long long st = (long long)gridDim.x * blockDim.x;
  for (; i < total; i += st) {
    int k = (int)(i % Kpad);
    long long m = i / Kpad;
    dst[i] = (k < K) ? src[m * K + k] : 0.0f;
  }
}

// Swizzle f32 weight (K x N) into WMMA-fragment-ordered bf16:
// dst[(((t*NK + ks)*32 + lane)*16 + i)] where t = n/16 tile, NK = Kpad/32.
// Fragment element i (pair p=i/2, hi=i&1) maps to
//   k = ks*32 + (p/4)*16 + (lane/16)*8 + (p%4)*2 + hi,  n = t*16 + (lane%16)
__global__ void k_swz_b(const float* __restrict__ src, unsigned short* __restrict__ dst,
                        int K, int N, int Kpad) {
  const int NK = Kpad >> 5;
  long long total = (long long)Kpad * N;
  long long i = (long long)blockIdx.x * blockDim.x + threadIdx.x;
  long long st = (long long)gridDim.x * blockDim.x;
  for (; i < total; i += st) {
    int ii = (int)(i & 15);
    long long r = i >> 4;
    int lane = (int)(r & 31); r >>= 5;
    int ks = (int)(r % NK);
    int t  = (int)(r / NK);
    int p = ii >> 1, hi = ii & 1;
    int k = (ks << 5) + ((p >> 2) << 4) + ((lane >> 4) << 3) + ((p & 3) << 1) + hi;
    int n = (t << 4) + (lane & 15);
    float v = (k < K) ? src[(long long)k * N + n] : 0.0f;
    dst[i] = f32_to_bf16_rne(v);
  }
}

__global__ void k_bcast_vn(const float* __restrict__ src, float* __restrict__ dst) {
  int i = blockIdx.x * blockDim.x + threadIdx.x;
  if (i < B_ * H_) dst[i] = src[i & (H_ - 1)];
}

// ---------------- WMMA bf16 GEMM ----------------
// C[M x N] = act( BN( (alpha*A + A2) @ B + bias ) )
// Block = 128 threads = 4 waves; wave w computes m-tile (blockIdx.y*4+w), all waves
// share one 64-column strip (blockIdx.x). For KPAD<=256 the strip's pre-swizzled
// B fragments (contiguous!) are staged once into LDS and read via ds_load_b128.
// grid = (N/64, M/64), block = 128.
template<int KPAD, bool HAS_A2>
__global__ void k_wmma_gemm(const float* __restrict__ A, const float* __restrict__ A2,
                            const float* __restrict__ epsP, int lda,
                            const unsigned short* __restrict__ Bsw,
                            const float* __restrict__ bias,
                            const float* __restrict__ bng, const float* __restrict__ bnb,
                            int relu, float* __restrict__ C, int ldc) {
  constexpr int NK = KPAD / 32;
  constexpr bool STAGE = (KPAD <= 256);
  constexpr int SBELEMS = STAGE ? (NK * 4 * 512) : 8;   // 4 tiles * 32 lanes * 16 bf16 per kstep
  __shared__ unsigned short sB[SBELEMS];

  const int lane = threadIdx.x & 31;
  const int wave = threadIdx.x >> 5;
  const int j  = blockIdx.x;                      // 64-column strip
  const int m0 = ((blockIdx.y << 2) + wave) << 4; // 16-row tile per wave
  const int mr = lane & 15;
  const int kh = lane >> 4;

  if constexpr (STAGE) {
    // strip's fragments are one contiguous block of NK*4*512 elems = NK*256 uint4
    const uint4* bsrc = (const uint4*)(Bsw + (size_t)(j * 4 * NK) * 512);
    uint4* bdst = (uint4*)sB;
#pragma unroll
    for (int it = 0; it < NK * 2; ++it)
      bdst[it * 128 + threadIdx.x] = bsrc[it * 128 + threadIdx.x];
    __syncthreads();
  }

  float alpha = 1.0f;
  if (HAS_A2) alpha = 1.0f + epsP[0];
  const float* arow  = A + (size_t)(m0 + mr) * lda;
  const float* arow2 = HAS_A2 ? (A2 + (size_t)(m0 + mr) * lda) : nullptr;

  v8f acc[4];
#pragma unroll
  for (int tt = 0; tt < 4; ++tt) acc[tt] = (v8f){0.f,0.f,0.f,0.f,0.f,0.f,0.f,0.f};

#pragma unroll 4
  for (int ks = 0; ks < NK; ++ks) {
    const int kb = (ks << 5) + (kh << 3);
    float4 a0 = *(const float4*)(arow + kb);
    float4 a1 = *(const float4*)(arow + kb + 4);
    float4 a2 = *(const float4*)(arow + kb + 16);
    float4 a3 = *(const float4*)(arow + kb + 20);
    if (HAS_A2) {
      float4 c0 = *(const float4*)(arow2 + kb);
      float4 c1 = *(const float4*)(arow2 + kb + 4);
      float4 c2 = *(const float4*)(arow2 + kb + 16);
      float4 c3 = *(const float4*)(arow2 + kb + 20);
      a0.x = fmaf(alpha, a0.x, c0.x); a0.y = fmaf(alpha, a0.y, c0.y);
      a0.z = fmaf(alpha, a0.z, c0.z); a0.w = fmaf(alpha, a0.w, c0.w);
      a1.x = fmaf(alpha, a1.x, c1.x); a1.y = fmaf(alpha, a1.y, c1.y);
      a1.z = fmaf(alpha, a1.z, c1.z); a1.w = fmaf(alpha, a1.w, c1.w);
      a2.x = fmaf(alpha, a2.x, c2.x); a2.y = fmaf(alpha, a2.y, c2.y);
      a2.z = fmaf(alpha, a2.z, c2.z); a2.w = fmaf(alpha, a2.w, c2.w);
      a3.x = fmaf(alpha, a3.x, c3.x); a3.y = fmaf(alpha, a3.y, c3.y);
      a3.z = fmaf(alpha, a3.z, c3.z); a3.w = fmaf(alpha, a3.w, c3.w);
    }
    v16bf av;   // native fptrunc (RNE) -> hardware bf16 convert
    av[0]  = (__bf16)a0.x; av[1]  = (__bf16)a0.y; av[2]  = (__bf16)a0.z; av[3]  = (__bf16)a0.w;
    av[4]  = (__bf16)a1.x; av[5]  = (__bf16)a1.y; av[6]  = (__bf16)a1.z; av[7]  = (__bf16)a1.w;
    av[8]  = (__bf16)a2.x; av[9]  = (__bf16)a2.y; av[10] = (__bf16)a2.z; av[11] = (__bf16)a2.w;
    av[12] = (__bf16)a3.x; av[13] = (__bf16)a3.y; av[14] = (__bf16)a3.z; av[15] = (__bf16)a3.w;
    if constexpr (STAGE) {
#pragma unroll
      for (int tt = 0; tt < 4; ++tt) {
        const v16bf bv = *(const v16bf*)(sB + ((size_t)(tt * NK + ks) * 32 + lane) * 16);
        acc[tt] = __builtin_amdgcn_wmma_f32_16x16x32_bf16(false, av, false, bv,
                                                          (short)0, acc[tt], false, false);
      }
    } else {
#pragma unroll
      for (int tt = 0; tt < 4; ++tt) {
        const int t = (j << 2) + tt;
        const v16bf bv = *(const v16bf*)(Bsw + ((size_t)(t * NK + ks) * 32 + lane) * 16);
        acc[tt] = __builtin_amdgcn_wmma_f32_16x16x32_bf16(false, av, false, bv,
                                                          (short)0, acc[tt], false, false);
      }
    }
  }

  const float rsq = rsqrtf(1.0f + 1e-5f);
#pragma unroll
  for (int tt = 0; tt < 4; ++tt) {
    int n = (j << 6) + (tt << 4) + mr;
    float bs = bias ? bias[n] : 0.0f;
    float sc = bng ? bng[n] * rsq : 1.0f;
    float sh = bng ? bnb[n] : 0.0f;
#pragma unroll
    for (int r = 0; r < 8; ++r) {
      int m = m0 + r + (kh << 3);
      float v = acc[tt][r] + bs;
      v = v * sc + sh;
      if (relu) v = fmaxf(v, 0.0f);
      C[(size_t)m * ldc + n] = v;
    }
  }
}

// ---------------- message passing: gather + relu + scatter-add (float4) ----------------
__global__ void k_scatter(const float* __restrict__ h, const int* __restrict__ e_index,
                          const int* __restrict__ e_type, const float* __restrict__ edge_emb,
                          float* __restrict__ agg) {
  long long total = (long long)B_ * E_ * (H_ / 4);
  long long i = (long long)blockIdx.x * blockDim.x + threadIdx.x;
  long long st = (long long)gridDim.x * blockDim.x;
  for (; i < total; i += st) {
    int f = (int)(i & 31) << 2;
    long long r = i >> 5;
    int e = (int)(r % E_);
    int b = (int)(r / E_);
    int src = e_index[(long long)b * 2 * E_ + e];
    int dst = e_index[(long long)b * 2 * E_ + E_ + e];
    int et  = e_type[(long long)b * E_ + e];
    float4 hv = *(const float4*)&h[((long long)b * N_ + src) * H_ + f];
    float4 ee = *(const float4*)&edge_emb[et * H_ + f];
    float* ap = &agg[((long long)b * N_ + dst) * H_ + f];
    atomicAdd(ap + 0, fmaxf(hv.x + ee.x, 0.0f));
    atomicAdd(ap + 1, fmaxf(hv.y + ee.y, 0.0f));
    atomicAdd(ap + 2, fmaxf(hv.z + ee.z, 0.0f));
    atomicAdd(ap + 3, fmaxf(hv.w + ee.w, 0.0f));
  }
}

// ---------------- BN2 + mask + residual + LayerNorm + vn node-sum ----------------
__global__ void k_postmlp(const float* __restrict__ z2, const float* __restrict__ h,
                          const unsigned char* __restrict__ mask,
                          const float* __restrict__ bn2g, const float* __restrict__ bn2b,
                          const float* __restrict__ lng, const float* __restrict__ lnb,
                          float* __restrict__ hln, float* __restrict__ nsum) {
  int node = blockIdx.x;
  int f = threadIdx.x;
  int b = node >> 10;
  float mk = mask[node] ? 1.0f : 0.0f;
  const float rsq = rsqrtf(1.0f + 1e-5f);
  long long base = (long long)node * H_;
  float x = h[base + f] + (z2[base + f] * (bn2g[f] * rsq) + bn2b[f]) * mk;
  __shared__ float s1[H_], s2[H_];
  s1[f] = x; s2[f] = x * x;
  __syncthreads();
  for (int s = 64; s > 0; s >>= 1) {
    if (f < s) { s1[f] += s1[f + s]; s2[f] += s2[f + s]; }
    __syncthreads();
  }
  float m = s1[0] * (1.0f / H_);
  float var = s2[0] * (1.0f / H_) - m * m;
  float y = (x - m) * rsqrtf(var + 1e-5f) * lng[f] + lnb[f];
  hln[base + f] = y;
  atomicAdd(&nsum[b * H_ + f], y * mk);
}

// ---------------- virtual node MLP ----------------
__global__ void k_vn_mlp(float* __restrict__ vn, const float* __restrict__ nsum,
                         const float* __restrict__ W1, const float* __restrict__ b1,
                         const float* __restrict__ g1, const float* __restrict__ bb1,
                         const float* __restrict__ W2, const float* __restrict__ b2,
                         const float* __restrict__ g2, const float* __restrict__ bb2) {
  int b = blockIdx.x;
  int f = threadIdx.x;
  __shared__ float sin_[H_], smid[H_];
  const float rsq = rsqrtf(1.0f + 1e-5f);
  float vold = vn[b * H_ + f];
  float vnew = vold + nsum[b * H_ + f];
  sin_[f] = vnew;
  __syncthreads();
  float acc = b1[f];
  for (int k = 0; k < H_; ++k) acc += sin_[k] * W1[k * H_ + f];
  acc = fmaxf(acc * (g1[f] * rsq) + bb1[f], 0.0f);
  smid[f] = acc;
  __syncthreads();
  float acc2 = b2[f];
  for (int k = 0; k < H_; ++k) acc2 += smid[k] * W2[k * H_ + f];
  vn[b * H_ + f] = (acc2 * (g2[f] * rsq) + bb2[f]) + vold;
}

// ---------------- gated update + JK running column-sum ----------------
__global__ void k_update_h(const float* __restrict__ hln, const float* __restrict__ vn,
                           const float* __restrict__ gateP, const unsigned char* __restrict__ mask,
                           float* __restrict__ h, float* __restrict__ jk, int slot) {
  float gt = 1.0f / (1.0f + __expf(-gateP[0]));
  long long total = (long long)B_ * N_ * H_;
  long long i = (long long)blockIdx.x * blockDim.x + threadIdx.x;
  long long st = (long long)gridDim.x * blockDim.x;
  for (; i < total; i += st) {
    int f = (int)(i & (H_ - 1));
    long long node = i >> 7;
    int b = (int)(node >> 10);
    float mk = mask[node] ? 1.0f : 0.0f;
    float v = (hln[i] + gt * vn[b * H_ + f]) * mk;
    h[i] = v;
    atomicAdd(&jk[(long long)b * (H_ * (L_ + 1)) + slot * H_ + f], v);
  }
}

__global__ void k_accum_jk(const float* __restrict__ h, const unsigned char* __restrict__ mask,
                           float* __restrict__ jk, int slot) {
  long long total = (long long)B_ * N_ * H_;
  long long i = (long long)blockIdx.x * blockDim.x + threadIdx.x;
  long long st = (long long)gridDim.x * blockDim.x;
  for (; i < total; i += st) {
    int f = (int)(i & (H_ - 1));
    long long node = i >> 7;
    int b = (int)(node >> 10);
    float mk = mask[node] ? 1.0f : 0.0f;
    atomicAdd(&jk[(long long)b * (H_ * (L_ + 1)) + slot * H_ + f], h[i] * mk);
  }
}

__global__ void k_concat2(const float* __restrict__ g, const float* __restrict__ f2,
                          float* __restrict__ comb) {
  int i = blockIdx.x * blockDim.x + threadIdx.x;
  if (i >= B_ * 2 * FUS_) return;
  int b = i / (2 * FUS_), j = i % (2 * FUS_);
  comb[i] = (j < FUS_) ? g[b * FUS_ + j] : f2[b * FUS_ + (j - FUS_)];
}

__global__ void k_cl_head(const float* __restrict__ z, const float* __restrict__ W,
                          const float* __restrict__ b, float* __restrict__ out) {
  int idx = blockIdx.x * blockDim.x + threadIdx.x;
  if (idx >= B_ * NC_) return;
  int bb = idx / NC_, c = idx % NC_;
  float acc = b[c];
  for (int k = 0; k < H_; ++k) acc += z[bb * H_ + k] * W[k * NC_ + c];
  out[idx] = acc;
}

// ---------------- host orchestration ----------------
extern "C" void kernel_launch(void* const* d_in, const int* in_sizes, int n_in,
                              void* d_out, int out_size, void* d_ws, size_t ws_size,
                              hipStream_t stream) {
  (void)in_sizes; (void)n_in; (void)out_size; (void)ws_size;
  const float* x_node   = (const float*)d_in[0];
  const int*   e_index  = (const int*)d_in[1];
  const int*   e_type   = (const int*)d_in[2];
  const unsigned char* mask = (const unsigned char*)d_in[3];
  const float* hand     = (const float*)d_in[4];
  const float* enc_W  = (const float*)d_in[5];
  const float* enc_b  = (const float*)d_in[6];
  const float* enc_g  = (const float*)d_in[7];
  const float* enc_bb = (const float*)d_in[8];
  const float* edge_emb = (const float*)d_in[9];
  const float* epsA   = (const float*)d_in[10];
  const float* gW1  = (const float*)d_in[11];
  const float* gb1  = (const float*)d_in[12];
  const float* gg1  = (const float*)d_in[13];
  const float* gbb1 = (const float*)d_in[14];
  const float* gW2  = (const float*)d_in[15];
  const float* gb2  = (const float*)d_in[16];
  const float* gg2  = (const float*)d_in[17];
  const float* gbb2 = (const float*)d_in[18];
  const float* lng  = (const float*)d_in[19];
  const float* lnb  = (const float*)d_in[20];
  const float* vn_init = (const float*)d_in[21];
  const float* vW1  = (const float*)d_in[22];
  const float* vb1  = (const float*)d_in[23];
  const float* vg1  = (const float*)d_in[24];
  const float* vbb1 = (const float*)d_in[25];
  const float* vW2  = (const float*)d_in[26];
  const float* vb2  = (const float*)d_in[27];
  const float* vg2  = (const float*)d_in[28];
  const float* vbb2 = (const float*)d_in[29];
  const float* vgate = (const float*)d_in[30];
  const float* gpW  = (const float*)d_in[31];
  const float* gpb  = (const float*)d_in[32];
  const float* gpg  = (const float*)d_in[33];
  const float* gpbb = (const float*)d_in[34];
  const float* feW1  = (const float*)d_in[35];
  const float* feb1  = (const float*)d_in[36];
  const float* feg1  = (const float*)d_in[37];
  const float* febb1 = (const float*)d_in[38];
  const float* feW2  = (const float*)d_in[39];
  const float* feb2  = (const float*)d_in[40];
  const float* feg2  = (const float*)d_in[41];
  const float* febb2 = (const float*)d_in[42];
  const float* clW1 = (const float*)d_in[43];
  const float* clb1 = (const float*)d_in[44];
  const float* clg  = (const float*)d_in[45];
  const float* clbb = (const float*)d_in[46];
  const float* clW2 = (const float*)d_in[47];
  const float* clb2 = (const float*)d_in[48];

  char* ws = (char*)d_ws;
  const size_t BIG = (size_t)B_ * N_ * H_ * sizeof(float);   // 33,554,432 B
  float* bufA = (float*)(ws);               // h
  float* bufB = (float*)(ws + BIG);         // z1 / h_ln
  float* bufC = (float*)(ws + 2 * BIG);     // agg / z2
  size_t off = 3 * BIG;
  auto take = [&](size_t bytes) -> void* {
    void* p = ws + off;
    off += (bytes + 255) & ~(size_t)255;
    return p;
  };
  float* xpad = (float*)take((size_t)B_ * N_ * 64 * 4);      // padded node feats (65536 x 64)
  float* hpad = (float*)take((size_t)B_ * 224 * 4);          // padded handcrafted (64 x 224)
  unsigned short* wenc = (unsigned short*)take((size_t)64 * H_ * 2);
  unsigned short* wg1  = (unsigned short*)take((size_t)L_ * H_ * H_ * 2);
  unsigned short* wg2  = (unsigned short*)take((size_t)L_ * H_ * H_ * 2);
  unsigned short* wgp  = (unsigned short*)take((size_t)(H_ * (L_ + 1)) * FUS_ * 2);
  unsigned short* wfe1 = (unsigned short*)take((size_t)224 * FUS_ * 2);
  unsigned short* wfe2 = (unsigned short*)take((size_t)FUS_ * FUS_ * 2);
  unsigned short* wcl1 = (unsigned short*)take((size_t)(2 * FUS_) * H_ * 2);
  float* vn   = (float*)take((size_t)B_ * H_ * 4);
  float* nsum = (float*)take((size_t)B_ * H_ * 4);
  float* jk   = (float*)take((size_t)B_ * H_ * (L_ + 1) * 4);
  float* gbuf = (float*)take((size_t)B_ * FUS_ * 4);
  float* f1   = (float*)take((size_t)B_ * FUS_ * 4);
  float* f2   = (float*)take((size_t)B_ * FUS_ * 4);
  float* comb = (float*)take((size_t)B_ * 2 * FUS_ * 4);
  float* zcl  = (float*)take((size_t)B_ * H_ * 4);

  hipStream_t s = stream;

  // pad odd-K activations; swizzle all weights into WMMA fragment order (bf16)
  k_pad_rows<<<4096, 256, 0, s>>>(x_node, xpad, FD_, 64, (long long)B_ * N_);
  k_pad_rows<<<64,   256, 0, s>>>(hand, hpad, HC_, 224, (long long)B_);
  k_swz_b<<<64,  256, 0, s>>>(enc_W, wenc, FD_, H_, 64);
  for (int l = 0; l < L_; ++l) {
    k_swz_b<<<128, 256, 0, s>>>(gW1 + (size_t)l * H_ * H_, wg1 + (size_t)l * H_ * H_, H_, H_, H_);
    k_swz_b<<<128, 256, 0, s>>>(gW2 + (size_t)l * H_ * H_, wg2 + (size_t)l * H_ * H_, H_, H_, H_);
  }
  k_swz_b<<<512, 256, 0, s>>>(gpW, wgp, H_ * (L_ + 1), FUS_, H_ * (L_ + 1));
  k_swz_b<<<256, 256, 0, s>>>(feW1, wfe1, HC_, FUS_, 224);
  k_swz_b<<<256, 256, 0, s>>>(feW2, wfe2, FUS_, FUS_, FUS_);
  k_swz_b<<<256, 256, 0, s>>>(clW1, wcl1, 2 * FUS_, H_, 2 * FUS_);

  // node encoder: h = relu(BN(X @ W + b))
  dim3 gE(H_ / 64, (B_ * N_) / 64);   // (2, 1024), 128 threads = 4 waves
  k_wmma_gemm<64, false><<<gE, 128, 0, s>>>(xpad, nullptr, nullptr, 64, wenc,
                                            enc_b, enc_g, enc_bb, 1, bufA, H_);

  k_zero<<<64, 256, 0, s>>>(jk, (long long)B_ * H_ * (L_ + 1));
  k_accum_jk<<<16384, 256, 0, s>>>(bufA, mask, jk, 0);
  k_bcast_vn<<<32, 256, 0, s>>>(vn_init, vn);

  for (int l = 0; l < L_; ++l) {
    k_zero<<<4096, 256, 0, s>>>(bufC, (long long)B_ * N_ * H_);
    k_zero<<<32, 256, 0, s>>>(nsum, (long long)B_ * H_);
    k_scatter<<<16384, 256, 0, s>>>(bufA, e_index, e_type, edge_emb, bufC);
    // z1 = relu(BN1( ((1+eps)h + agg) @ W1 + b1 ))
    k_wmma_gemm<128, true><<<gE, 128, 0, s>>>(bufA, bufC, epsA + l, H_,
                                              wg1 + (size_t)l * H_ * H_,
                                              gb1 + l * H_, gg1 + l * H_, gbb1 + l * H_, 1,
                                              bufB, H_);
    // z2 = z1 @ W2 + b2
    k_wmma_gemm<128, false><<<gE, 128, 0, s>>>(bufB, nullptr, nullptr, H_,
                                               wg2 + (size_t)l * H_ * H_,
                                               gb2 + l * H_, nullptr, nullptr, 0,
                                               bufC, H_);
    k_postmlp<<<B_ * N_, H_, 0, s>>>(bufC, bufA, mask,
                                     gg2 + l * H_, gbb2 + l * H_,
                                     lng + l * H_, lnb + l * H_, bufB, nsum);
    k_vn_mlp<<<B_, H_, 0, s>>>(vn, nsum,
                               vW1 + (size_t)l * H_ * H_, vb1 + l * H_, vg1 + l * H_, vbb1 + l * H_,
                               vW2 + (size_t)l * H_ * H_, vb2 + l * H_, vg2 + l * H_, vbb2 + l * H_);
    k_update_h<<<16384, 256, 0, s>>>(bufB, vn, vgate + l, mask, bufA, jk, l + 1);
  }

  // readout head (M = 64 -> one block-row of 4 waves)
  dim3 gGP(FUS_ / 64, 1);             // (4, 1)
  k_wmma_gemm<768, false><<<gGP, 128, 0, s>>>(jk, nullptr, nullptr, H_ * (L_ + 1), wgp,
                                              gpb, gpg, gpbb, 1, gbuf, FUS_);
  k_wmma_gemm<224, false><<<gGP, 128, 0, s>>>(hpad, nullptr, nullptr, 224, wfe1,
                                              feb1, feg1, febb1, 1, f1, FUS_);
  k_wmma_gemm<256, false><<<gGP, 128, 0, s>>>(f1, nullptr, nullptr, FUS_, wfe2,
                                              feb2, feg2, febb2, 1, f2, FUS_);
  k_concat2<<<128, 256, 0, s>>>(gbuf, f2, comb);
  dim3 gCL(H_ / 64, 1);               // (2, 1)
  k_wmma_gemm<512, false><<<gCL, 128, 0, s>>>(comb, nullptr, nullptr, 2 * FUS_, wcl1,
                                              clb1, clg, clbb, 1, zcl, H_);
  k_cl_head<<<(B_ * NC_ + 255) / 256, 256, 0, s>>>(zcl, clW2, clb2, (float*)d_out);
}
